// EfficientAttention_44959717654720
// MI455X (gfx1250) — matrix operand order repaired and checked by reading, not compiled, not run on hardware
//
#include <hip/hip_runtime.h>
#include <hip/hip_bf16.h>

// ---------------------------------------------------------------------------
// EfficientAttention for MI455X (gfx1250, wave32, WMMA 16x16x32 f16).
// B=8, N=4096, C=768, H=12, D=64.  ~161 GFLOP dominated by two dense GEMMs.
// All matrix ops via v_wmma_f32_16x16x32_f16 (f32 accumulate).  Tiles staged
// into double-buffered LDS with GLOBAL_LOAD_ASYNC_TO_LDS_B128 (ASYNCcnt),
// B operands pre-transposed so every staging copy is a contiguous 16B chunk.
// ---------------------------------------------------------------------------

typedef __attribute__((ext_vector_type(16))) _Float16 v16h;
typedef __attribute__((ext_vector_type(8)))  _Float16 v8h;
typedef __attribute__((ext_vector_type(8)))  float    v8f;
typedef __attribute__((ext_vector_type(4)))  int      v4i;

#define B_   8
#define N_   4096
#define C_   768
#define H_   12
#define D_   64
#define C3_  (3 * C_)

// ------------------------------------------------ async global->LDS staging
#if __has_builtin(__builtin_amdgcn_global_load_async_to_lds_b128)
#define HAVE_ASYNC_LDS 1
typedef __attribute__((address_space(1))) v4i* as1_v4i_ptr;
typedef __attribute__((address_space(3))) v4i* as3_v4i_ptr;
__device__ __forceinline__ void copy16_to_lds(const _Float16* g, _Float16* l) {
    __builtin_amdgcn_global_load_async_to_lds_b128(
        (as1_v4i_ptr)g, (as3_v4i_ptr)l, 0, 0);
}
#else
#define HAVE_ASYNC_LDS 0
__device__ __forceinline__ void copy16_to_lds(const _Float16* g, _Float16* l) {
    *reinterpret_cast<uint4*>(l) = *reinterpret_cast<const uint4*>(g);
}
#endif

__device__ __forceinline__ void wait_async_lds() {
#if HAVE_ASYNC_LDS
#if __has_builtin(__builtin_amdgcn_s_wait_asynccnt)
    __builtin_amdgcn_s_wait_asynccnt(0);
#else
    asm volatile("s_wait_asynccnt 0x0" ::: "memory");
#endif
#endif
}

// ----------------------------------------------------------------- fp32->fp16
__global__ void cvt_f32_to_f16(const float* __restrict__ in,
                               _Float16* __restrict__ out, size_t n) {
    size_t i = (size_t)blockIdx.x * blockDim.x + threadIdx.x;
    size_t stride = (size_t)gridDim.x * blockDim.x;
    for (; i < n; i += stride) out[i] = (_Float16)in[i];
}

// --------------------------------------- fp32 [K x Ntot] -> fp16 Bt [Ntot x K]
__global__ void cvt_transpose_f16(const float* __restrict__ in,
                                  _Float16* __restrict__ out,
                                  int K, int Ntot) {
    size_t idx = (size_t)blockIdx.x * blockDim.x + threadIdx.x;
    const size_t total = (size_t)K * Ntot;
    const size_t stride = (size_t)gridDim.x * blockDim.x;
    for (; idx < total; idx += stride) {
        const size_t n = idx / K, k = idx % K;
        out[idx] = (_Float16)in[k * Ntot + n];
    }
}

// ------------------------------- extract v^T [bh*D + e][n] from qkv_f16 (f16)
__global__ __launch_bounds__(256)
void extract_vT(const _Float16* __restrict__ qkv, _Float16* __restrict__ vT) {
    const int row = blockIdx.x;           // bh*64 + e
    const int bh = row >> 6, e = row & 63;
    const int b = bh / H_, h = bh % H_;
    const _Float16* src = qkv + (size_t)b * N_ * C3_ + 2 * C_ + h * D_ + e;
    _Float16* dst = vT + (size_t)row * N_;
    for (int n = threadIdx.x; n < N_; n += 256)
        dst[n] = src[(size_t)n * C3_];
}

// ----------------------------------------------------- column softmax over N
// One block per (b, h, d) column of the k section of qkv_f16; writes k^T f16.
__global__ __launch_bounds__(256)
void softmax_k_cols(const _Float16* __restrict__ qkv,
                    _Float16* __restrict__ kT) {
    __shared__ float red[256];
    const int id = blockIdx.x;             // 0 .. B*H*D-1, ordered (b, h, d)
    const int t  = threadIdx.x;
    const int b  = id / (H_ * D_);
    const int hd = id % (H_ * D_);
    const _Float16* col = qkv + (size_t)b * N_ * C3_ + C_ + hd;

    float vals[16];
    float m = -1e30f;
#pragma unroll
    for (int i = 0; i < 16; ++i) {
        const int n = t + i * 256;
        vals[i] = (float)col[(size_t)n * C3_];
        m = fmaxf(m, vals[i]);
    }
    red[t] = m; __syncthreads();
    for (int s = 128; s > 0; s >>= 1) {
        if (t < s) red[t] = fmaxf(red[t], red[t + s]);
        __syncthreads();
    }
    m = red[0]; __syncthreads();

    float sum = 0.f;
#pragma unroll
    for (int i = 0; i < 16; ++i) { vals[i] = __expf(vals[i] - m); sum += vals[i]; }
    red[t] = sum; __syncthreads();
    for (int s = 128; s > 0; s >>= 1) {
        if (t < s) red[t] += red[t + s];
        __syncthreads();
    }
    const float inv = 1.f / red[0];

    _Float16* dst = kT + (size_t)id * N_;
#pragma unroll
    for (int i = 0; i < 16; ++i)
        dst[t + i * 256] = (_Float16)(vals[i] * inv);
}

// --------------------------------------------------------- generic WMMA GEMM
// C[M x Ntot] = A[M x K] @ B[K x Ntot], B given pre-transposed Bt[Ntot x K].
// Block tile: BM x 64, BK=32, double-buffered LDS, async global->LDS staging.
// Wave w owns rows [w*16, w*16+16) and all four 16-col tiles (4 accumulators).
// Batch offsets decompose as (z / Hdiv, z % Hdiv) so per-head slices of the
// interleaved qkv buffer need no relayout.  CT=true writes C transposed.
template <int BM>
__device__ __forceinline__ void stage_tiles(const _Float16* Ablk,
                                            const _Float16* Btblk,
                                            int lda, int ldbt, int kk,
                                            _Float16* lA, _Float16* lB, int t) {
    constexpr int CA = BM * 4;           // A chunks (8 halves each)
    constexpr int TOT = CA + 256;        // + B chunks (64 rows x 4)
#pragma unroll
    for (int c = t; c < TOT; c += BM * 2) {
        if (c < CA) {
            const int row = c >> 2, col = (c & 3) << 3;
            copy16_to_lds(Ablk + (size_t)row * lda + kk + col,
                          &lA[row * 32 + col]);
        } else {
            const int cb = c - CA;
            const int n = cb >> 2, col = (cb & 3) << 3;
            copy16_to_lds(Btblk + (size_t)n * ldbt + kk + col,
                          &lB[n * 32 + col]);
        }
    }
}

template <int BM, bool OUT_F16, bool BIAS, bool CT>
__global__ __launch_bounds__(BM * 2)
void wmma_gemm_kernel(const _Float16* __restrict__ A,
                      const _Float16* __restrict__ Bt,
                      void* __restrict__ Cout,
                      const float* __restrict__ bias,
                      int K, int lda, int ldbt, int ldc, int Hdiv,
                      long long sAo, long long sAi,
                      long long sBo, long long sBi,
                      long long sCo, long long sCi) {
    constexpr int BN = 64, BK = 32;
    __shared__ _Float16 lA[2][BM * BK];     // [row][k]
    __shared__ _Float16 lB[2][BN * BK];     // [n][k]

    const int t  = threadIdx.x;
    const int z  = blockIdx.z;
    const long long zo = z / Hdiv, zi = z % Hdiv;
    const _Float16* Ablk  = A  + zo * sAo + zi * sAi + (size_t)blockIdx.x * BM * lda;
    const _Float16* Btblk = Bt + zo * sBo + zi * sBi + (size_t)blockIdx.y * BN * ldbt;
    const long long offC = zo * sCo + zi * sCi;

    const int lane = t & 31;
    const int wv   = t >> 5;
    const int m16  = lane & 15;     // column within 16x16 tile (B/C/D layouts)
    const int kh   = lane >> 4;     // half-wave selector (A/B layouts)

    v8f acc[4];
#pragma unroll
    for (int i = 0; i < 4; ++i) acc[i] = (v8f){0.f,0.f,0.f,0.f,0.f,0.f,0.f,0.f};

    // prologue: stage first tile pair
    stage_tiles<BM>(Ablk, Btblk, lda, ldbt, 0, lA[0], lB[0], t);
    wait_async_lds();
    __syncthreads();

    int buf = 0;
    for (int kk = 0; kk < K; kk += BK) {
        // issue async staging of the next tile pair into the other buffer
        if (kk + BK < K)
            stage_tiles<BM>(Ablk, Btblk, lda, ldbt, kk + BK,
                            lA[buf ^ 1], lB[buf ^ 1], t);
        if (kk + 2 * BK < K)
            __builtin_prefetch(Ablk + (size_t)(t & (BM - 1)) * lda + kk + 2 * BK, 0, 0);

        // ---- A fragment: 16x32 f16 layout (two ds_load_b128 per lane) ------
        const v8h* pa = reinterpret_cast<const v8h*>(
            &lA[buf][(wv * 16 + m16) * BK + kh * 8]);
        const v8h a0 = pa[0];
        const v8h a1 = pa[2];                     // +16 halves
        const v16h af = __builtin_shufflevector(a0, a1,
            0, 1, 2, 3, 4, 5, 6, 7, 8, 9, 10, 11, 12, 13, 14, 15);

        // ---- 4 x (B fragment + WMMA) --------------------------------------
#pragma unroll
        for (int tc = 0; tc < 4; ++tc) {
            const v8h* pb = reinterpret_cast<const v8h*>(
                &lB[buf][(tc * 16 + m16) * BK + kh * 8]);
            const v8h b0 = pb[0];
            const v8h b1 = pb[2];
            const v16h bf = __builtin_shufflevector(b0, b1,
                0, 1, 2, 3, 4, 5, 6, 7, 8, 9, 10, 11, 12, 13, 14, 15);
            acc[tc] = __builtin_amdgcn_wmma_f32_16x16x32_f16(
                false, af, false, bf, (short)0, acc[tc], false, false);
        }

        wait_async_lds();       // next tile resident in LDS
        __syncthreads();        // everyone done reading `buf`
        buf ^= 1;
    }

    // ---- writeback: D layout -> VGPR r holds row (r + 8*kh), col m16 -------
    const long long crow0 = (long long)blockIdx.x * BM + wv * 16 + kh * 8;
#pragma unroll
    for (int tc = 0; tc < 4; ++tc) {
        const long long gcol = (long long)blockIdx.y * BN + tc * 16 + m16;
        float bv = 0.f;
        if constexpr (BIAS) bv = bias[gcol];
#pragma unroll
        for (int r = 0; r < 8; ++r) {
            const long long grow = crow0 + r;
            const long long idx = CT ? offC + gcol * (long long)ldc + grow
                                     : offC + grow * (long long)ldc + gcol;
            if constexpr (OUT_F16)
                ((_Float16*)Cout)[idx] = (_Float16)acc[tc][r];
            else
                ((float*)Cout)[idx] = acc[tc][r] + bv;
        }
    }
}

// ---------------------------------------------------------------------------
extern "C" void kernel_launch(void* const* d_in, const int* in_sizes, int n_in,
                              void* d_out, int out_size, void* d_ws, size_t ws_size,
                              hipStream_t stream) {
    const float* x     = (const float*)d_in[0];
    const float* Wqkv  = (const float*)d_in[1];
    const float* Wproj = (const float*)d_in[2];
    const float* bproj = (const float*)d_in[3];
    float* out = (float*)d_out;

    const size_t szX    = (size_t)B_ * N_ * C_;
    const size_t szWq   = (size_t)C_ * C3_;
    const size_t szWp   = (size_t)C_ * C_;
    const size_t szQKV  = (size_t)B_ * N_ * C3_;
    const size_t szKT   = (size_t)B_ * H_ * D_ * N_;
    const size_t szVT   = szKT;
    const size_t szCtx  = (size_t)B_ * H_ * D_ * D_;
    const size_t szAttn = szX;

    char* p = (char*)d_ws;
    _Float16* x_h    = (_Float16*)p; p += szX    * 2;
    _Float16* wqT_h  = (_Float16*)p; p += szWq   * 2;   // [3C x C]
    _Float16* wpT_h  = (_Float16*)p; p += szWp   * 2;   // [C x C] transposed
    _Float16* qkv_h  = (_Float16*)p; p += szQKV  * 2;
    _Float16* kT_h   = (_Float16*)p; p += szKT   * 2;   // [bh*D + d][n]
    _Float16* vT_h   = (_Float16*)p; p += szVT   * 2;   // [bh*D + e][n]
    _Float16* ctxT_h = (_Float16*)p; p += szCtx  * 2;   // [bh][e][d]
    _Float16* attn_h = (_Float16*)p; p += szAttn * 2;

    // 1) precision conversion (+ transpose for B operands)
    cvt_f32_to_f16<<<4096, 256, 0, stream>>>(x, x_h, szX);
    cvt_transpose_f16<<<2048, 256, 0, stream>>>(Wqkv,  wqT_h, C_, C3_);
    cvt_transpose_f16<<<1024, 256, 0, stream>>>(Wproj, wpT_h, C_, C_);

    // 2) qkv = x @ Wqkv   (M=32768, K=768, N=2304) -> f16
    {
        dim3 g((B_ * N_) / 128, C3_ / 64, 1);
        wmma_gemm_kernel<128, true, false, false><<<g, 256, 0, stream>>>(
            x_h, wqT_h, qkv_h, nullptr,
            /*K*/ C_, /*lda*/ C_, /*ldbt*/ C_, /*ldc*/ C3_, /*Hdiv*/ H_,
            0, 0, 0, 0, 0, 0);
    }

    // 3) softmax over sequence on k -> k^T ; extract v^T
    softmax_k_cols<<<B_ * H_ * D_, 256, 0, stream>>>(qkv_h, kT_h);
    extract_vT<<<B_ * H_ * D_, 256, 0, stream>>>(qkv_h, vT_h);

    // 4) ctx^T[bh] = (k^T[bh] (64 x 4096) @ v[bh] (4096 x 64))^T, via CT write
    {
        dim3 g(1, 1, B_ * H_);
        wmma_gemm_kernel<64, true, false, true><<<g, 128, 0, stream>>>(
            kT_h, vT_h, ctxT_h, nullptr,
            /*K*/ N_, /*lda*/ N_, /*ldbt*/ N_, /*ldc*/ D_, /*Hdiv*/ H_,
            /*sAo*/ (long long)H_ * D_ * N_, /*sAi*/ (long long)D_ * N_,
            /*sBo*/ (long long)H_ * D_ * N_, /*sBi*/ (long long)D_ * N_,
            /*sCo*/ (long long)H_ * D_ * D_, /*sCi*/ (long long)D_ * D_);
    }

    // 5) attn[bh] = q[bh] (4096 x 64) @ ctx[bh] (64 x 64); q read in place,
    //    Bt = ctx^T; written directly in [b, n, h*D+e] layout
    {
        dim3 g(N_ / 128, 1, B_ * H_);
        wmma_gemm_kernel<128, true, false, false><<<g, 256, 0, stream>>>(
            qkv_h, ctxT_h, attn_h, nullptr,
            /*K*/ D_, /*lda*/ C3_, /*ldbt*/ D_, /*ldc*/ C_, /*Hdiv*/ H_,
            /*sAo*/ (long long)N_ * C3_,     /*sAi*/ (long long)D_,
            /*sBo*/ (long long)H_ * D_ * D_, /*sBi*/ (long long)D_ * D_,
            /*sCo*/ (long long)N_ * C_,      /*sCi*/ (long long)D_);
    }

    // 6) out = attn @ Wproj + bproj   (M=32768, K=768, N=768) -> f32
    {
        dim3 g((B_ * N_) / 128, C_ / 64, 1);
        wmma_gemm_kernel<128, false, true, false><<<g, 256, 0, stream>>>(
            attn_h, wpT_h, (void*)out, bproj,
            /*K*/ C_, /*lda*/ C_, /*ldbt*/ C_, /*ldc*/ C_, /*Hdiv*/ H_,
            0, 0, 0, 0, 0, 0);
    }
}
